// TextModel_15341623182074
// MI455X (gfx1250) — matrix operand order repaired and checked by reading, not compiled, not run on hardware
//
#include <hip/hip_runtime.h>
#include <hip/hip_bf16.h>
#include <stdint.h>

#define VOCAB_ 32000
#define EMB_   1024
#define UNITS_ 1024
#define B_     16
#define S_     256

typedef __attribute__((ext_vector_type(16))) __bf16 v16bf;
typedef __attribute__((ext_vector_type(8)))  __bf16 v8bf;
typedef __attribute__((ext_vector_type(8)))  float  v8f;

// ---------------- WMMA helpers (CDNA5 16x16x32 bf16, f32 accum) ----------------

static __device__ inline v8f wmma_bf16(v16bf a, v16bf b, v8f c) {
  // (neg_a, A, neg_b, B, c_mod, C, reuse_a, reuse_b)
  return __builtin_amdgcn_wmma_f32_16x16x32_bf16(false, a, false, b, (short)0, c,
                                                 false, false);
}

// A-operand 16x32 bf16: lane holds row M=lane%16.
// half 0 (lanes 0-15):  K = kbase+{0..15}
// half 1 (lanes 16-31): K = kbase+{16..31}
static __device__ inline v16bf load_a_frag(const __bf16* row, int kbase, int hf) {
  v8bf lo = *(const v8bf*)(row + kbase + 16 * hf);
  v8bf hi = *(const v8bf*)(row + kbase + 16 * hf + 8);
  v16bf a;
#pragma unroll
  for (int i = 0; i < 8; ++i) { a[i] = lo[i]; a[i + 8] = hi[i]; }
  return a;
}

// B-operand 32x16 bf16: lane holds column N=lane%16 of B = row N of B^T.
// half 0: K = kbase+{0..15};  half 1: K = kbase+{16..31}  (contiguous 32B)
static __device__ inline v16bf load_b_frag(const __bf16* row, int kbase, int hf) {
  return *(const v16bf*)(row + kbase + 16 * hf);
}

static __device__ inline v16bf load_b_frag_f32(const float* row, int kbase, int hf) {
  const float* q = row + kbase + 16 * hf;
  v16bf b;
#pragma unroll
  for (int i = 0; i < 16; ++i) b[i] = (__bf16)q[i];
  return b;
}

// ---------------- Tensor Data Mover (gfx1250): 1D contiguous tile -> LDS ----------------

#if __has_builtin(__builtin_amdgcn_tensor_load_to_lds)
#define HAVE_TDM 1
typedef unsigned int v4u __attribute__((ext_vector_type(4)));
typedef int v8i __attribute__((ext_vector_type(4 * 2)));
typedef int v4i __attribute__((ext_vector_type(4)));
typedef __attribute__((address_space(3))) char lds_char_t;

static __device__ inline unsigned lds_addr_of(const void* p) {
  // generic -> LDS addrspace cast; numeric value of an AS3 pointer is the LDS byte offset
  return (unsigned)(unsigned long long)(const lds_char_t*)p;
}

// Load `n2` 2-byte elements (contiguous) from global `gptr` into LDS offset `lds_off`.
// D# Group0: count=1, lds_addr, global_addr[56:0], type=2 ("image").
// D# Group1: data_size=1 (2B), tensor_dim0=n2, tile_dim0=n2, tile_dim1=1, dim0_stride=n2.
static __device__ inline void tdm_load_1d_b16(unsigned lds_off, const void* gptr,
                                              unsigned n2) {
  const unsigned long long ga = (unsigned long long)(uintptr_t)gptr;
  v4u g0 = {1u,                                   // count=1, no gather, not restore
            lds_off,                              // lds_addr
            (unsigned)(ga & 0xffffffffu),         // global_addr[31:0]
            (unsigned)((ga >> 32) & 0x1ffffffu) | (2u << 30)};  // addr[56:32] | type=2
  v8i g1 = {(int)(1u << 16),                      // wg_mask=0, data_size=1 (2 bytes)
            (int)((n2 & 0xffffu) << 16),          // tensor_dim0[15:0] (bits 63:48)
            (int)(((n2 >> 16) & 0xffffu) | (1u << 16)),  // tensor_dim0[31:16] | tensor_dim1=1
            (int)((n2 & 0xffffu) << 16),          // tensor_dim1[31:16]=0 | tile_dim0
            1,                                    // tile_dim1=1, tile_dim2=0
            (int)n2,                              // tensor_dim0_stride[31:0]
            0, 0};                                // stride0[47:32], dim1_stride
  v4i z4 = {0, 0, 0, 0};
#if defined(__clang_major__) && __clang_major__ >= 23
  v8i z8 = {0, 0, 0, 0, 0, 0, 0, 0};
  __builtin_amdgcn_tensor_load_to_lds(g0, g1, z4, z4, z8, 0);
#else
  __builtin_amdgcn_tensor_load_to_lds(g0, g1, z4, z4, 0);
#endif
}
#else
#define HAVE_TDM 0
#endif

// ---------------- elementwise cast + tiled transpose-cast ----------------

__global__ void k_cast_bf16(__bf16* __restrict__ dst, const float* __restrict__ src, int n) {
  int i = blockIdx.x * blockDim.x + threadIdx.x;
  if (i < n) dst[i] = (__bf16)src[i];
}

// src: f32 [R][C]  ->  dst: bf16 [C][R]
__global__ void k_transpose_bf16(__bf16* __restrict__ dst, const float* __restrict__ src,
                                 int R, int C) {
  __shared__ float tile[32][33];
  const int c0 = blockIdx.x * 32, r0 = blockIdx.y * 32;
  for (int i = threadIdx.y; i < 32; i += 8)
    tile[i][threadIdx.x] = src[(size_t)(r0 + i) * C + (c0 + threadIdx.x)];
  __syncthreads();
  for (int i = threadIdx.y; i < 32; i += 8)
    dst[(size_t)(c0 + i) * R + (r0 + threadIdx.x)] = (__bf16)tile[threadIdx.x][i];
}

// ---------------- X = gather(E, tokens) @ w_in^T  -> bf16 [S*16][1024] ----------------

__global__ void __launch_bounds__(256) k_embed_gemm(
    __bf16* __restrict__ Xbf, const int* __restrict__ tokens,
    const float* __restrict__ emb, const float* __restrict__ w_in) {
  const int t = blockIdx.x;      // timestep
  const int nblk = blockIdx.y;   // 8 blocks of 128 output cols
  __shared__ __align__(16) __bf16 As[16 * 1024];  // gathered emb rows, bf16
  const int tid = threadIdx.x;
  {
    const int b = tid >> 4;
    const int c0 = (tid & 15) * 64;
    const int tok = tokens[b * S_ + t];
    const float* src = emb + (size_t)tok * EMB_ + c0;
    for (int j = 0; j < 64; ++j) As[b * 1024 + c0 + j] = (__bf16)src[j];
  }
  __syncthreads();
  const int wv = tid >> 5, lane = tid & 31, hf = lane >> 4, l16 = lane & 15;
  const int ntile = nblk * 8 + wv;
  const int n = ntile * 16 + l16;
  const float* brow = w_in + (size_t)n * EMB_;  // w_in[n][k] is already B^T layout
  const __bf16* arow = As + l16 * 1024;
  v8f acc = {};
  for (int kk = 0; kk < 1024; kk += 32)
    acc = wmma_bf16(load_a_frag(arow, kk, hf), load_b_frag_f32(brow, kk, hf), acc);
#pragma unroll
  for (int r = 0; r < 8; ++r) {
    const int b = r + 8 * hf;
    Xbf[(size_t)(t * 16 + b) * 1024 + n] = (__bf16)acc[r];
  }
}

// ---------------- C = A*A (1024^3), writes C and C^T (both bf16) ----------------

__global__ void k_square(__bf16* __restrict__ Crow, __bf16* __restrict__ CT,
                         const __bf16* __restrict__ Arow, const __bf16* __restrict__ AT) {
  const int bx = blockIdx.x, by = blockIdx.y;  // n-tile, m-tile
  const int lane = threadIdx.x & 31, hf = lane >> 4, l16 = lane & 15;
  const __bf16* arow = Arow + (size_t)(by * 16 + l16) * 1024;
  const __bf16* bt = AT + (size_t)(bx * 16 + l16) * 1024;
  v8f acc = {};
  for (int kk = 0; kk < 1024; kk += 32)
    acc = wmma_bf16(load_a_frag(arow, kk, hf), load_b_frag(bt, kk, hf), acc);
#pragma unroll
  for (int r = 0; r < 8; ++r) {
    const int m = by * 16 + r + 8 * hf, n = bx * 16 + l16;
    const __bf16 v = (__bf16)acc[r];
    Crow[(size_t)m * 1024 + n] = v;
    CT[(size_t)n * 1024 + m] = v;
  }
}

// ---------------- sequential within-group pass (mode 0: b_g, mode 1: final) ----------------

__global__ void __launch_bounds__(1024) k_group_seq(
    float* __restrict__ GrpMem, const __bf16* __restrict__ Xbf,
    const __bf16* __restrict__ WT, const float* __restrict__ Sg,
    __bf16* __restrict__ Membf, const float* __restrict__ memory0, int mode) {
  const int g = blockIdx.x;  // 16 groups of 16 steps
  float* mem = GrpMem + (size_t)g * 16 * 1024;
  __shared__ __align__(16) __bf16 Abf[16 * 1024];
  const int tid = threadIdx.x;
  for (int j = 0; j < 16; ++j) {
    const int i = tid * 16 + j;
    float init = 0.f;
    if (g == 0) init = memory0[i];                      // exact initial state
    else if (mode == 1) init = Sg[(size_t)(g - 1) * 16 * 1024 + i];
    mem[i] = init;
  }
  __syncthreads();
  const int wv = tid >> 5, lane = tid & 31, hf = lane >> 4, l16 = lane & 15;
  for (int step = 0; step < 16; ++step) {
    const int t = g * 16 + step;
    for (int j = 0; j < 16; ++j) {
      const int i = tid * 16 + j;
      Abf[i] = (__bf16)(mem[i] + (float)Xbf[(size_t)t * 16 * 1024 + i]);
    }
    __syncthreads();
    float nv[2][8];
#pragma unroll
    for (int s = 0; s < 2; ++s) {
      const int n = (wv * 2 + s) * 16 + l16;
      const __bf16* bt = WT + (size_t)n * 1024;
      const __bf16* arow = Abf + l16 * 1024;
      v8f acc = {};
      for (int kk = 0; kk < 1024; kk += 32)
        acc = wmma_bf16(load_a_frag(arow, kk, hf), load_b_frag(bt, kk, hf), acc);
#pragma unroll
      for (int r = 0; r < 8; ++r) nv[s][r] = acc[r];
    }
#pragma unroll
    for (int s = 0; s < 2; ++s) {
      const int n = (wv * 2 + s) * 16 + l16;
#pragma unroll
      for (int r = 0; r < 8; ++r) {
        const int row = r + 8 * hf;
        mem[(size_t)row * 1024 + n] = nv[s][r];
        if (mode == 1) Membf[(size_t)(t * 16 + row) * 1024 + n] = (__bf16)nv[s][r];
      }
    }
    __syncthreads();
  }
  // mode 0: GrpMem[g] now holds b_g
}

// ---------------- group-level scan: s_g = s_{g-1} @ W^16 + b_g ----------------

__global__ void __launch_bounds__(1024) k_scan(
    float* __restrict__ sstate, float* __restrict__ Sg,
    const float* __restrict__ Bg, const __bf16* __restrict__ W16T) {
  __shared__ __align__(16) __bf16 Abf[16 * 1024];
  const int tid = threadIdx.x;
  for (int j = 0; j < 16; ++j) sstate[tid * 16 + j] = 0.f;
  __syncthreads();
  const int wv = tid >> 5, lane = tid & 31, hf = lane >> 4, l16 = lane & 15;
  for (int g = 0; g < 16; ++g) {
    for (int j = 0; j < 16; ++j) { const int i = tid * 16 + j; Abf[i] = (__bf16)sstate[i]; }
    __syncthreads();
#pragma unroll
    for (int s = 0; s < 2; ++s) {
      const int n = (wv * 2 + s) * 16 + l16;
      const __bf16* bt = W16T + (size_t)n * 1024;
      const __bf16* arow = Abf + l16 * 1024;
      v8f acc = {};
      for (int kk = 0; kk < 1024; kk += 32)
        acc = wmma_bf16(load_a_frag(arow, kk, hf), load_b_frag(bt, kk, hf), acc);
#pragma unroll
      for (int r = 0; r < 8; ++r) {
        const int row = r + 8 * hf;
        const float v = acc[r] + Bg[(size_t)g * 16 * 1024 + (size_t)row * 1024 + n];
        sstate[(size_t)row * 1024 + n] = v;
        Sg[(size_t)g * 16 * 1024 + (size_t)row * 1024 + n] = v;
      }
    }
    __syncthreads();
  }
}

// ---------------- classifier GEMM + bias + per-row max / exp-sum (per V-split) ----------------

__global__ void __launch_bounds__(1024) k_cls_gemm(
    float* __restrict__ out, const __bf16* __restrict__ Membf,
    const __bf16* __restrict__ WclsT, const float* __restrict__ bcls,
    float* __restrict__ rmax4, float* __restrict__ rsum4) {
  const int vs = blockIdx.x;  // 0..3 (V split into 4 x 8000)
  const int tp = blockIdx.y;  // 0..127 (pairs of timesteps -> 32 rows)
  const int t0 = tp * 2;
  __shared__ __align__(16) __bf16 Abf[32 * 1024];  // 64 KB, reused for reductions
  const int tid = threadIdx.x;
  // Stage the contiguous 64 KB A-tile via the Tensor Data Mover (one wave issues the
  // DMA, waits on TENSORcnt, then the workgroup barrier publishes the LDS tile).
#if HAVE_TDM
  if (tid < 32) {
    tdm_load_1d_b16(lds_addr_of(Abf), Membf + (size_t)t0 * 16 * 1024, 32u * 1024u);
    __builtin_amdgcn_s_wait_tensorcnt(0);
  }
#else
  {
    const __bf16* src = Membf + (size_t)t0 * 16 * 1024;
#pragma unroll
    for (int j = 0; j < 32; ++j) Abf[j * 1024 + tid] = src[j * 1024 + tid];
  }
#endif
  __syncthreads();
  const int wv = tid >> 5, lane = tid & 31, hf = lane >> 4, l16 = lane & 15;
  const size_t SV = (size_t)S_ * VOCAB_;
  float pmax[2][8];
#pragma unroll
  for (int mt = 0; mt < 2; ++mt)
#pragma unroll
    for (int r = 0; r < 8; ++r) pmax[mt][r] = -1e30f;
  for (int nti = wv; nti < 500; nti += 32) {
    const int v = (vs * 500 + nti) * 16 + l16;
    const __bf16* brow = WclsT + (size_t)v * 1024;
    // next B row this wave will consume (clamped): prefetch while computing
    const int vnx = (nti + 32 < 500) ? v + 512 : v;
    const __bf16* brow_nx = WclsT + (size_t)vnx * 1024;
    const __bf16* a0 = Abf + l16 * 1024;
    const __bf16* a1 = Abf + (16 + l16) * 1024;
    v8f acc0 = {}, acc1 = {};
    for (int kk = 0; kk < 1024; kk += 32) {
      __builtin_prefetch(brow_nx + kk + 16 * hf, 0, 1);  // global_prefetch_b8
      v16bf bf = load_b_frag(brow, kk, hf);  // B frag shared by both M-tiles
      acc0 = wmma_bf16(load_a_frag(a0, kk, hf), bf, acc0);
      acc1 = wmma_bf16(load_a_frag(a1, kk, hf), bf, acc1);
    }
    const float bias = bcls[v];
#pragma unroll
    for (int r = 0; r < 8; ++r) {
      const int b = r + 8 * hf;
      const float l0 = acc0[r] + bias, l1 = acc1[r] + bias;
      pmax[0][r] = fmaxf(pmax[0][r], l0);
      pmax[1][r] = fmaxf(pmax[1][r], l1);
      out[(size_t)b * SV + (size_t)t0 * VOCAB_ + v] = l0;
      out[(size_t)b * SV + (size_t)(t0 + 1) * VOCAB_ + v] = l1;
    }
  }
#pragma unroll
  for (int off = 1; off <= 8; off <<= 1)
#pragma unroll
    for (int mt = 0; mt < 2; ++mt)
#pragma unroll
      for (int r = 0; r < 8; ++r)
        pmax[mt][r] = fmaxf(pmax[mt][r], __shfl_xor(pmax[mt][r], off, 32));
  __syncthreads();  // all Abf reads done -> safe to alias for reductions
  float* red = (float*)Abf;       // [32 waves][32 rows]
  float* brmax = red + 32 * 32;   // [32]
  if (l16 == 0) {
#pragma unroll
    for (int mt = 0; mt < 2; ++mt)
#pragma unroll
      for (int r = 0; r < 8; ++r) red[wv * 32 + mt * 16 + r + 8 * hf] = pmax[mt][r];
  }
  __syncthreads();
  if (tid < 32) {
    float m = -1e30f;
    for (int w = 0; w < 32; ++w) m = fmaxf(m, red[w * 32 + tid]);
    brmax[tid] = m;
  }
  __syncthreads();
  // pass B: sum of exp(logit - rowmax), re-reading L2-hot logits
  float psum[2][8];
#pragma unroll
  for (int mt = 0; mt < 2; ++mt)
#pragma unroll
    for (int r = 0; r < 8; ++r) psum[mt][r] = 0.f;
  for (int nti = wv; nti < 500; nti += 32) {
    const int v = (vs * 500 + nti) * 16 + l16;
#pragma unroll
    for (int r = 0; r < 8; ++r) {
      const int b = r + 8 * hf;
      const float x0 = out[(size_t)b * SV + (size_t)t0 * VOCAB_ + v];
      const float x1 = out[(size_t)b * SV + (size_t)(t0 + 1) * VOCAB_ + v];
      psum[0][r] += __expf(x0 - brmax[b]);
      psum[1][r] += __expf(x1 - brmax[16 + b]);
    }
  }
#pragma unroll
  for (int off = 1; off <= 8; off <<= 1)
#pragma unroll
    for (int mt = 0; mt < 2; ++mt)
#pragma unroll
      for (int r = 0; r < 8; ++r) psum[mt][r] += __shfl_xor(psum[mt][r], off, 32);
  if (l16 == 0) {
#pragma unroll
    for (int mt = 0; mt < 2; ++mt)
#pragma unroll
      for (int r = 0; r < 8; ++r) red[wv * 32 + mt * 16 + r + 8 * hf] = psum[mt][r];
  }
  __syncthreads();
  if (tid < 32) {
    float s = 0.f;
    for (int w = 0; w < 32; ++w) s += red[w * 32 + tid];
    const int rid = t0 * 16 + tid;  // row id = t*16 + b
    rmax4[rid * 4 + vs] = brmax[tid];
    rsum4[rid * 4 + vs] = s;
  }
}

// ---------------- combine V-splits, then normalize ----------------

__global__ void k_combine(float* __restrict__ rowm, float* __restrict__ rowz,
                          const float* __restrict__ rmax4, const float* __restrict__ rsum4) {
  const int rid = blockIdx.x * blockDim.x + threadIdx.x;
  if (rid >= 4096) return;
  float m = -1e30f;
#pragma unroll
  for (int vs = 0; vs < 4; ++vs) m = fmaxf(m, rmax4[rid * 4 + vs]);
  float Z = 0.f;
#pragma unroll
  for (int vs = 0; vs < 4; ++vs) Z += rsum4[rid * 4 + vs] * __expf(rmax4[rid * 4 + vs] - m);
  rowm[rid] = m;
  rowz[rid] = 1.0f / Z;
}

__global__ void k_norm(float* __restrict__ out, const float* __restrict__ rowm,
                       const float* __restrict__ rowz) {
  const int v = blockIdx.x * 256 + threadIdx.x;
  const int t = blockIdx.y, b = blockIdx.z;
  const int rid = t * 16 + b;
  const size_t idx = (size_t)b * S_ * VOCAB_ + (size_t)t * VOCAB_ + v;
  out[idx] = __expf(out[idx] - rowm[rid]) * rowz[rid];
}

// ---------------- host launch ----------------

extern "C" void kernel_launch(void* const* d_in, const int* in_sizes, int n_in,
                              void* d_out, int out_size, void* d_ws, size_t ws_size,
                              hipStream_t stream) {
  const int* tokens = (const int*)d_in[0];
  const float* memory0 = (const float*)d_in[1];
  const float* emb = (const float*)d_in[2];
  const float* w_in = (const float*)d_in[3];
  const float* w_ctx = (const float*)d_in[4];
  const float* w_cls = (const float*)d_in[5];
  const float* b_cls = (const float*)d_in[6];
  float* out = (float*)d_out;

  char* p = (char*)d_ws;
  auto take = [&](size_t bytes) {
    char* r = p;
    p += (bytes + 255) & ~(size_t)255;
    return (void*)r;
  };
  __bf16* Wbf = (__bf16*)take((size_t)1024 * 1024 * 2);
  __bf16* WctxT = (__bf16*)take((size_t)1024 * 1024 * 2);
  __bf16* Qa_row = (__bf16*)take((size_t)1024 * 1024 * 2);
  __bf16* Qa_T = (__bf16*)take((size_t)1024 * 1024 * 2);
  __bf16* Qb_row = (__bf16*)take((size_t)1024 * 1024 * 2);
  __bf16* Qb_T = (__bf16*)take((size_t)1024 * 1024 * 2);
  __bf16* WclsT = (__bf16*)take((size_t)VOCAB_ * 1024 * 2);
  __bf16* Xbf = (__bf16*)take((size_t)4096 * 1024 * 2);
  __bf16* Membf = (__bf16*)take((size_t)4096 * 1024 * 2);
  float* GrpMem = (float*)take((size_t)16 * 16 * 1024 * 4);
  float* Sg = (float*)take((size_t)16 * 16 * 1024 * 4);
  float* sstate = (float*)take((size_t)16 * 1024 * 4);
  float* rmax4 = (float*)take((size_t)4096 * 4 * 4);
  float* rsum4 = (float*)take((size_t)4096 * 4 * 4);
  float* rowm = (float*)take((size_t)4096 * 4);
  float* rowz = (float*)take((size_t)4096 * 4);

  // weight prep: bf16 copies + transposes (B-operand wants row-contiguous columns)
  k_cast_bf16<<<4096, 256, 0, stream>>>(Wbf, w_ctx, 1024 * 1024);
  k_transpose_bf16<<<dim3(32, 32), dim3(32, 8), 0, stream>>>(WctxT, w_ctx, 1024, 1024);
  k_transpose_bf16<<<dim3(1000, 32), dim3(32, 8), 0, stream>>>(WclsT, w_cls, 1024, VOCAB_);
  // X_t = E[tok_t] @ w_in^T, all steps in parallel
  k_embed_gemm<<<dim3(S_, 8), 256, 0, stream>>>(Xbf, tokens, emb, w_in);
  // W^16 via repeated squaring (each pass emits both layouts)
  k_square<<<dim3(64, 64), 32, 0, stream>>>(Qa_row, Qa_T, Wbf, WctxT);    // W^2
  k_square<<<dim3(64, 64), 32, 0, stream>>>(Qb_row, Qb_T, Qa_row, Qa_T);  // W^4
  k_square<<<dim3(64, 64), 32, 0, stream>>>(Qa_row, Qa_T, Qb_row, Qb_T);  // W^8
  k_square<<<dim3(64, 64), 32, 0, stream>>>(Qb_row, Qb_T, Qa_row, Qa_T);  // W^16
  // chunked affine scan of the recurrence (serial depth 256 -> ~48)
  k_group_seq<<<16, 1024, 0, stream>>>(GrpMem, Xbf, WctxT, Sg, Membf, memory0, 0);
  k_scan<<<1, 1024, 0, stream>>>(sstate, Sg, GrpMem, Qb_T);
  k_group_seq<<<16, 1024, 0, stream>>>(GrpMem, Xbf, WctxT, Sg, Membf, memory0, 1);
  // classifier GEMM + softmax statistics, combine, normalize
  k_cls_gemm<<<dim3(4, 128), 1024, 0, stream>>>(out, Membf, WclsT, b_cls, rmax4, rsum4);
  k_combine<<<16, 256, 0, stream>>>(rowm, rowz, rmax4, rsum4);
  k_norm<<<dim3(125, S_, B_), 256, 0, stream>>>(out, rowm, rowz);

  (void)in_sizes; (void)n_in; (void)out_size; (void)ws_size;
}
